// Globally_Attentive_6090263626194
// MI455X (gfx1250) — matrix operand order repaired and checked by reading, not compiled, not run on hardware
//
#include <hip/hip_runtime.h>
#include <hip/hip_bf16.h>

typedef __attribute__((ext_vector_type(16))) __bf16 v16bf;
typedef __attribute__((ext_vector_type(8)))  float  v8f;

__device__ __forceinline__ v16bf ld_frag(const __bf16* row, int off0, int off1) {
    union { v16bf v; uint4 q[2]; } f;
    f.q[0] = *(const uint4*)(row + off0);
    f.q[1] = *(const uint4*)(row + off1);
    return f.v;
}

__device__ __forceinline__ v8f wmma_bf16(v16bf a, v16bf b, v8f c) {
    // D = A(16x32) * B(32x16) + C, f32 accum
    return __builtin_amdgcn_wmma_f32_16x16x32_bf16(false, a, false, b, (short)0, c, false, false);
}

// ---- CDNA5 async global->LDS (ASYNCcnt-tracked), inline asm per ISA ch.10 ----
__device__ __forceinline__ unsigned lds_off(const void* p) {
    // generic LDS pointer: addr[31:0] is the LDS byte offset (ISA aperture rules)
    return (unsigned)(uintptr_t)p;
}
__device__ __forceinline__ void async_copy_b128(unsigned loff, const void* g) {
    asm volatile("global_load_async_to_lds_b128 %0, %1, off" :: "v"(loff), "v"(g) : "memory");
}
__device__ __forceinline__ void async_copy_b32(unsigned loff, const void* g) {
    asm volatile("global_load_async_to_lds_b32 %0, %1, off" :: "v"(loff), "v"(g) : "memory");
}
__device__ __forceinline__ void wait_async0() {
    asm volatile("s_wait_asynccnt 0x0" ::: "memory");
}

// -------------------------------------------------------------------------
// Kernel 1: h[b][s][n] = LN1(x transposed)[s][n] + PE[s][n]   (f32)
// x layout (B,N,K,S): batch b = b_*128 + k_; h row s contiguous in n.
// -------------------------------------------------------------------------
__global__ __launch_bounds__(256) void k1_ln_pe(const float* __restrict__ xg,
                                                const float* __restrict__ w1,
                                                const float* __restrict__ b1,
                                                float* __restrict__ h) {
    extern __shared__ char smem[];
    float* tile = (float*)smem;           // 256 x 65 (padded)
    float* Mu   = tile + 256 * 65;        // 64
    float* Rs   = Mu + 64;                // 64

    const int b = blockIdx.x, t = threadIdx.x;
    const int b_ = b >> 7, k_ = b & 127;
    const long xbase = (((long)b_ * 256) * 128 + k_) * 256;
    const long hbase = (long)b * 65536;
    const float wln = w1[t], bln = b1[t];
    const float freq = __expf(-0.03597789f * (float)(t & ~1));
    const unsigned tileOff = lds_off(tile);

    for (int c = 0; c < 4; ++c) {
        const int s0 = c * 64;
        #pragma unroll 4
        for (int i = 0; i < 64; ++i) {
            int idx = t + i * 256;
            int n = idx >> 6, sl = idx & 63;
            async_copy_b32(tileOff + (unsigned)(n * 65 + sl) * 4,
                           xg + xbase + (long)n * 32768 + s0 + sl);
        }
        wait_async0();
        __syncthreads();
        if (t < 64) {
            float s = 0.f, s2 = 0.f;
            for (int n = 0; n < 256; ++n) {
                float v = tile[n * 65 + t];
                s += v; s2 += v * v;
            }
            float mu = s * (1.f / 256.f);
            float var = s2 * (1.f / 256.f) - mu * mu;
            Mu[t] = mu; Rs[t] = rsqrtf(var + 1e-5f);
        }
        __syncthreads();
        #pragma unroll 4
        for (int i = 0; i < 64; ++i) {
            int s = s0 + i;
            float pe = (t & 1) ? __cosf((float)s * freq) : __sinf((float)s * freq);
            float v = (tile[t * 65 + i] - Mu[i]) * Rs[i] * wln + bln + pe;
            h[hbase + (long)s * 256 + t] = v;
        }
        __syncthreads();
    }
}

// -------------------------------------------------------------------------
// Kernel 2: {Q,K,V}[m][n] = h[m][:] . w[n][:] + bias[n]   bf16 out
// M = 512*256 rows; block = 32 rows x 256 cols; 8 waves of 16x64 slabs.
// -------------------------------------------------------------------------
__global__ __launch_bounds__(256) void k2_qkv(const float* __restrict__ h,
                                              const float* __restrict__ wq, const float* __restrict__ bq,
                                              const float* __restrict__ wk, const float* __restrict__ bk,
                                              const float* __restrict__ wv, const float* __restrict__ bv,
                                              __bf16* __restrict__ oq, __bf16* __restrict__ ok,
                                              __bf16* __restrict__ ov) {
    extern __shared__ char smem[];
    __bf16* Ab = (__bf16*)smem;                    // 32 x 256
    __bf16* Wl = (__bf16*)(smem + 16384);          // 256 x 256 (row n, col d)
    float*  Bl = (float*)(smem + 16384 + 131072);  // 256

    const float* w; const float* bb; __bf16* o;
    if (blockIdx.y == 0)      { w = wq; bb = bq; o = oq; }
    else if (blockIdx.y == 1) { w = wk; bb = bk; o = ok; }
    else                      { w = wv; bb = bv; o = ov; }

    const int t = threadIdx.x;
    const long m0 = (long)blockIdx.x * 32;

    #pragma unroll 4
    for (int i = 0; i < 32; ++i) { int idx = t + i * 256; Ab[idx] = (__bf16)h[m0 * 256 + idx]; }
    #pragma unroll 4
    for (int i = 0; i < 256; ++i){ int idx = t + i * 256; Wl[idx] = (__bf16)w[idx]; }
    Bl[t] = bb[t];
    __syncthreads();

    const int lane = t & 31, lo = lane & 15, kh = lane >> 4;
    const int wvid = t >> 5, mblk = wvid >> 2, nblk = wvid & 3;

    v8f acc[4] = {};
    const __bf16* arow = Ab + (16 * mblk + lo) * 256;
    #pragma unroll
    for (int ks = 0; ks < 256; ks += 32) {
        v16bf a = ld_frag(arow, ks + 8 * kh, ks + 16 + 8 * kh);
        v16bf bf[4];
        #pragma unroll
        for (int f = 0; f < 4; ++f) {
            int n = nblk * 64 + f * 16 + lo;
            bf[f] = ld_frag(Wl + n * 256, ks + 8 * kh, ks + 16 + 8 * kh);
        }
        #pragma unroll
        for (int f = 0; f < 4; ++f) acc[f] = wmma_bf16(a, bf[f], acc[f]);
    }
    #pragma unroll
    for (int f = 0; f < 4; ++f)
        #pragma unroll
        for (int r = 0; r < 8; ++r) {
            int m = 16 * mblk + r + 8 * kh;
            int n = nblk * 64 + f * 16 + lo;
            o[(m0 + m) * 256 + n] = (__bf16)(acc[f][r] + Bl[n]);
        }
}

// -------------------------------------------------------------------------
// Kernel 3 (per batch): att = softmax(Q K^T) * 1/16; h2 = res + att V;
// out = LN2(h2) transposed back + x.
// LDS: K(128K) + V^T(132K, stride 264) + Qblk/att(16K) + scores f32(32K) + stats
// K tile + Q blocks staged with global_load_async_to_lds_b128 (ASYNCcnt).
// -------------------------------------------------------------------------
__global__ __launch_bounds__(256) void k3_attn(const __bf16* __restrict__ qg,
                                               const __bf16* __restrict__ kg,
                                               const __bf16* __restrict__ vg,
                                               const float* __restrict__ hres,
                                               const float* __restrict__ xg,
                                               const float* __restrict__ ln2w,
                                               const float* __restrict__ ln2b,
                                               float* __restrict__ out) {
    extern __shared__ char smem[];
    __bf16* Kl = (__bf16*)smem;                               // 256 x 256
    __bf16* Vt = (__bf16*)(smem + 131072);                    // 256 x 264 (pad)
    __bf16* Qb = (__bf16*)(smem + 131072 + 135168);           // 32 x 256 (Q, then att)
    float*  Sc = (float*)(smem + 131072 + 135168 + 16384);    // 32 x 256
    float*  Ra = (float*)((char*)Sc + 32768);                 // 32
    float*  Rb = Ra + 32;                                     // 32

    const int b = blockIdx.x, t = threadIdx.x;
    const int lane = t & 31, lo = lane & 15, kh = lane >> 4;
    const int wvid = t >> 5, mblk = wvid >> 2, nblk = wvid & 3;
    const long bbase = (long)b * 65536;
    const float scale = 0.0625f;   // 1/sqrt(256), applied AFTER softmax per reference
    const unsigned klOff = lds_off(Kl), qbOff = lds_off(Qb);

    // ---- K tile: async copy (overlaps with V transpose below)
    {
        const char* src = (const char*)(kg + bbase);
        #pragma unroll 4
        for (int i = 0; i < 32; ++i) {
            int idx = t + i * 256;
            async_copy_b128(klOff + (unsigned)idx * 16, src + (long)idx * 16);
        }
    }
    // ---- V transposed into Vt[d][s] (padded rows: 264 elems = 528B, 16B aligned)
    for (int i = 0; i < 256; ++i)
        Vt[t * 264 + i] = vg[bbase + (long)i * 256 + t];
    // ---- Q block 0 prefetch
    {
        const char* src = (const char*)(qg + bbase);
        #pragma unroll
        for (int i = 0; i < 4; ++i) {
            int idx = t + i * 256;
            async_copy_b128(qbOff + (unsigned)idx * 16, src + (long)idx * 16);
        }
    }

    for (int q0 = 0; q0 < 256; q0 += 32) {
        wait_async0();        // this wave's async copies (K on first pass, Q block) done
        __syncthreads();      // publish to all waves; also guards Sc/Qb reuse

        // ---- scores = Qblk @ K^T  (B rows come from Kl rows: B[d][j] = K[j][d])
        {
            v8f acc[4] = {};
            const __bf16* arow = Qb + (16 * mblk + lo) * 256;
            #pragma unroll
            for (int ks = 0; ks < 256; ks += 32) {
                v16bf a = ld_frag(arow, ks + 8 * kh, ks + 16 + 8 * kh);
                v16bf bf[4];
                #pragma unroll
                for (int f = 0; f < 4; ++f) {
                    int j = nblk * 64 + f * 16 + lo;
                    bf[f] = ld_frag(Kl + j * 256, ks + 8 * kh, ks + 16 + 8 * kh);
                }
                #pragma unroll
                for (int f = 0; f < 4; ++f) acc[f] = wmma_bf16(a, bf[f], acc[f]);
            }
            #pragma unroll
            for (int f = 0; f < 4; ++f)
                #pragma unroll
                for (int r = 0; r < 8; ++r) {
                    int m = 16 * mblk + r + 8 * kh;
                    int j = nblk * 64 + f * 16 + lo;
                    Sc[m * 256 + j] = acc[f][r];
                }
        }
        __syncthreads();

        // ---- softmax row stats
        if (t < 32) {
            const float* r = Sc + t * 256;
            float m = -3.0e38f;
            for (int j = 0; j < 256; ++j) m = fmaxf(m, r[j]);
            float s = 0.f;
            for (int j = 0; j < 256; ++j) s += __expf(r[j] - m);
            Ra[t] = m; Rb[t] = 1.f / s;
        }
        __syncthreads();

        // ---- att (bf16) into Qb (Q dead); scale after softmax
        #pragma unroll 4
        for (int i = 0; i < 32; ++i) {
            int idx = t + i * 256;   // row i, col t
            Qb[idx] = (__bf16)(__expf(Sc[idx] - Ra[i]) * Rb[i] * scale);
        }
        __syncthreads();

        // ---- O = att @ V; h2 = O + residual -> Sc
        {
            v8f acc[4] = {};
            const __bf16* arow = Qb + (16 * mblk + lo) * 256;
            #pragma unroll
            for (int ks = 0; ks < 256; ks += 32) {
                v16bf a = ld_frag(arow, ks + 8 * kh, ks + 16 + 8 * kh);
                v16bf bf[4];
                #pragma unroll
                for (int f = 0; f < 4; ++f) {
                    int n = nblk * 64 + f * 16 + lo;
                    bf[f] = ld_frag(Vt + n * 264, ks + 8 * kh, ks + 16 + 8 * kh);
                }
                #pragma unroll
                for (int f = 0; f < 4; ++f) acc[f] = wmma_bf16(a, bf[f], acc[f]);
            }
            #pragma unroll
            for (int f = 0; f < 4; ++f)
                #pragma unroll
                for (int r = 0; r < 8; ++r) {
                    int m = 16 * mblk + r + 8 * kh;
                    int n = nblk * 64 + f * 16 + lo;
                    Sc[m * 256 + n] = acc[f][r] + hres[bbase + (long)(q0 + m) * 256 + n];
                }
        }
        __syncthreads();

        // ---- prefetch next Q block (att buffer is dead now); overlaps epilogue
        if (q0 + 32 < 256) {
            const char* src = (const char*)(qg + bbase + (long)(q0 + 32) * 256);
            #pragma unroll
            for (int i = 0; i < 4; ++i) {
                int idx = t + i * 256;
                async_copy_b128(qbOff + (unsigned)idx * 16, src + (long)idx * 16);
            }
        }

        // ---- LN2 row stats
        if (t < 32) {
            const float* r = Sc + t * 256;
            float s = 0.f, s2 = 0.f;
            for (int j = 0; j < 256; ++j) { float v = r[j]; s += v; s2 += v * v; }
            float mu = s * (1.f / 256.f);
            float var = s2 * (1.f / 256.f) - mu * mu;
            Ra[t] = mu; Rb[t] = rsqrtf(var + 1e-5f);
        }
        __syncthreads();

        // ---- out[b_][n][k_][q0+sl] = LN2(h2)[sl][n] + x  (s-contiguous stores)
        {
            const int b_ = b >> 7, k_ = b & 127;
            const int sl = lane;
            #pragma unroll 4
            for (int i = 0; i < 32; ++i) {
                int n = i * 8 + wvid;
                float y = (Sc[sl * 256 + n] - Ra[sl]) * Rb[sl] * ln2w[n] + ln2b[n];
                long o = (((long)b_ * 256 + n) * 128 + k_) * 256 + (q0 + sl);
                out[o] = y + xg[o];
            }
        }
        // loop-top __syncthreads() guards Sc/Qb reuse for next iteration
    }
}

// -------------------------------------------------------------------------
extern "C" void kernel_launch(void* const* d_in, const int* in_sizes, int n_in,
                              void* d_out, int out_size, void* d_ws, size_t ws_size,
                              hipStream_t stream) {
    const float* x    = (const float*)d_in[0];
    const float* ln1w = (const float*)d_in[1];
    const float* ln1b = (const float*)d_in[2];
    const float* qw   = (const float*)d_in[3];
    const float* qb   = (const float*)d_in[4];
    const float* kw   = (const float*)d_in[5];
    const float* kb   = (const float*)d_in[6];
    const float* vw   = (const float*)d_in[7];
    const float* vb   = (const float*)d_in[8];
    const float* ln2w = (const float*)d_in[9];
    const float* ln2b = (const float*)d_in[10];
    float* out = (float*)d_out;

    char* ws = (char*)d_ws;
    float*  h  = (float*)ws;                                  // 512*256*256 f32  = 128 MiB
    __bf16* q  = (__bf16*)(ws + 134217728);                   //  64 MiB
    __bf16* k  = (__bf16*)(ws + 134217728 + 67108864);        //  64 MiB
    __bf16* v  = (__bf16*)(ws + 134217728 + 2 * 67108864);    //  64 MiB

    const int LDS1 = 256 * 65 * 4 + 128 * 4;                  //  67072 B
    const int LDS2 = 16384 + 131072 + 1024;                   // 148480 B
    const int LDS3 = 131072 + 135168 + 16384 + 32768 + 256;   // 315648 B

    hipFuncSetAttribute((const void*)k1_ln_pe, hipFuncAttributeMaxDynamicSharedMemorySize, LDS1);
    hipFuncSetAttribute((const void*)k2_qkv,   hipFuncAttributeMaxDynamicSharedMemorySize, LDS2);
    hipFuncSetAttribute((const void*)k3_attn,  hipFuncAttributeMaxDynamicSharedMemorySize, LDS3);

    k1_ln_pe<<<512, 256, LDS1, stream>>>(x, ln1w, ln1b, h);
    k2_qkv<<<dim3(4096, 3), 256, LDS2, stream>>>(h, qw, qb, kw, kb, vw, vb, q, k, v);
    k3_attn<<<512, 256, LDS3, stream>>>(q, k, v, h, x, ln2w, ln2b, out);
}
